// SimpleTacotron_39402029973564
// MI455X (gfx1250) — compile-verified, matrix operand order checked by reading, not run on hardware
//
#include <hip/hip_runtime.h>

// ---------------------------------------------------------------------------
// SimpleTacotron forward for MI455X (gfx1250, wave32, WMMA f16 16x16x32).
//
//   1. embed gather -> x (f16, time-major rows); zero-padded f16 weights
//   2. GEMM (64x128 tile, async global->LDS staging); gate GEMMs store in
//      C-fragment-native layout so the scans read b128-contiguous per lane
//   3. encoder LSTM scan: both directions in one launch (2 blocks), h in LDS
//   4. dec_in (K padded to 96), q / Q / K / V projections, V transpose
//   5. fused flash attention (async K preload, transposed-V fragments)
//   6. out-projection, concat (time-major, padded to 608), decoder gate GEMM
//   7. decoder LSTM scan split across 8 WGs: h ping-pong in global memory,
//      one device-scope atomic barrier per step, async h->LDS copy
//   8. mel GEMM -> d_out (f32), stop dot kernel -> d_out tail
// ---------------------------------------------------------------------------

typedef _Float16 half_t;
typedef __attribute__((ext_vector_type(16))) _Float16 v16h;
typedef __attribute__((ext_vector_type(8)))  float    v8f;

#define Bc   32
#define Lc   256
#define Tc   800
#define Ec   256
#define Hc   256
#define Dc   512
#define Mc   80
#define ENCc 512
#define NHc  8
#define KQP  96    // dec_in K padded 80 -> 96
#define KDP  608   // lstm_in K padded 592 -> 608
#define DGW  8     // decoder scan workgroups

// CDNA5 16-bit A-matrix swizzle (ISA 7.12.2): fragment element t of a lane
// maps to K index ((t&8)<<1) | ((lane>=16)?8:0) | (t&7).
__device__ __forceinline__ int swz_k(int t, int lane) {
  return ((t & 8) << 1) | ((lane >> 4) << 3) | (t & 7);
}

__device__ __forceinline__ v16h load_frag_rm(const half_t* p0, int ld,
                                             int row0, int k0, int lane) {
  v16h v;
  const half_t* p = p0 + (size_t)(row0 + (lane & 15)) * ld + k0;
#pragma unroll
  for (int t = 0; t < 16; ++t) v[t] = p[swz_k(t, lane)];
  return v;
}

__device__ __forceinline__ v8f wmma_f16(v16h a, v16h b, v8f c) {
  return __builtin_amdgcn_wmma_f32_16x16x32_f16(false, a, false, b,
                                                (short)0, c, false, false);
}

// Async global -> LDS copy of 16 bytes per lane (CDNA5 ASYNCcnt path).
__device__ __forceinline__ void async_ldg_b128(void* lds, const void* g) {
  unsigned l = (unsigned)(uintptr_t)lds;   // LDS offset = addr[31:0]
  asm volatile("global_load_async_to_lds_b128 %0, %1, off"
               :: "v"(l), "v"(g) : "memory");
}
__device__ __forceinline__ void wait_async0() {
  asm volatile("s_wait_asynccnt 0x0" ::: "memory");
}

__device__ __forceinline__ float sigm(float x) { return 1.f / (1.f + __expf(-x)); }

// Device-wide sense barrier across G workgroups (agent-scope atomics).
__device__ __forceinline__ void gbar(unsigned* cnt, unsigned* gen, unsigned G) {
  __threadfence();
  __syncthreads();
  if (threadIdx.x == 0) {
    unsigned g0 = __hip_atomic_load(gen, __ATOMIC_RELAXED, __HIP_MEMORY_SCOPE_AGENT);
    unsigned old = __hip_atomic_fetch_add(cnt, 1u, __ATOMIC_ACQ_REL, __HIP_MEMORY_SCOPE_AGENT);
    if (old == G - 1u) {
      __hip_atomic_store(cnt, 0u, __ATOMIC_RELAXED, __HIP_MEMORY_SCOPE_AGENT);
      __hip_atomic_fetch_add(gen, 1u, __ATOMIC_RELEASE, __HIP_MEMORY_SCOPE_AGENT);
    } else {
      while (__hip_atomic_load(gen, __ATOMIC_ACQUIRE, __HIP_MEMORY_SCOPE_AGENT) == g0) {
        __builtin_amdgcn_s_sleep(1);
      }
    }
  }
  __syncthreads();
  __threadfence();
}

// ---------------------------------------------------------------------------
// Elementwise kernels
// ---------------------------------------------------------------------------
__global__ void k_zero(unsigned* __restrict__ p, int n) {
  int i = blockIdx.x * blockDim.x + threadIdx.x;
  if (i < n) p[i] = 0u;
}

// Cast f32 [R][K] to f16 [Rp][Kp], zero-filling the padding.
__global__ void k_cast_pad(const float* __restrict__ s, half_t* __restrict__ d,
                           int R, int K, int Rp, int Kp) {
  int i = blockIdx.x * blockDim.x + threadIdx.x;
  if (i >= Rp * Kp) return;
  int r = i / Kp, k = i % Kp;
  d[i] = (r < R && k < K) ? (half_t)s[(size_t)r * K + k] : (half_t)0.f;
}

// Embedding gather, written TIME-MAJOR: row = s*32 + b (so gate-GEMM M-tiles
// align with the scan's per-step fragment reads).
__global__ void k_embed(const int* __restrict__ text, const float* __restrict__ emb,
                        half_t* __restrict__ x) {
  int bs = blockIdx.x;              // b*L + s
  int e  = threadIdx.x;
  int b = bs / Lc, s = bs % Lc;
  int tok = text[bs];
  x[(size_t)(s * Bc + b) * Ec + e] = (half_t)emb[(size_t)tok * Ec + e];
}

// dec_in, rows [b*T+t], K-padded to 96 (cols 80..95 zero)
__global__ void k_decin(const float* __restrict__ mel, half_t* __restrict__ dec, int total) {
  int i = blockIdx.x * blockDim.x + threadIdx.x;
  if (i >= total) return;
  int bt = i / KQP, m = i % KQP;
  int t  = bt % Tc;
  dec[i] = (m < Mc && t > 0) ? (half_t)mel[(size_t)bt * Mc - Mc + m] : (half_t)0.f;
}

// lstm_in = [ctx(512), dec_in(80), zeros(16)], TIME-MAJOR rows (t*32+b),
// K-padded to 608.
__global__ void k_cat(const half_t* __restrict__ ctx, const half_t* __restrict__ dec,
                      half_t* __restrict__ out, int total) {
  int i = blockIdx.x * blockDim.x + threadIdx.x;
  if (i >= total) return;
  int c   = i % KDP;
  int bt2 = i / KDP;                  // t*32 + b
  int t = bt2 >> 5, b = bt2 & 31;
  size_t bt = (size_t)b * Tc + t;
  half_t v = (half_t)0.f;
  if (c < ENCc)            v = ctx[bt * ENCc + c];
  else if (c < ENCc + Mc)  v = dec[bt * KQP + (c - ENCc)];
  out[i] = v;
}

// V transpose for attention: Vt[(b*NH+h)][d][key] = Vp[(b*L+key)][h*64+d]
__global__ void k_vt(const half_t* __restrict__ Vp, half_t* __restrict__ Vt, int total) {
  int i = blockIdx.x * blockDim.x + threadIdx.x;
  if (i >= total) return;
  int key = i & (Lc - 1);
  int d   = (i >> 8) & 63;
  int bh  = i >> 14;
  int b = bh >> 3, hh = bh & 7;
  Vt[i] = Vp[((size_t)(b * Lc + key)) * ENCc + hh * 64 + d];
}

__global__ void k_stop(const half_t* __restrict__ h, const float* __restrict__ w,
                       const float* __restrict__ b, float* __restrict__ out) {
  int bt = blockIdx.x * blockDim.x + threadIdx.x;
  if (bt >= Bc * Tc) return;
  float acc = b[0];
  for (int d = 0; d < Dc; ++d) acc += (float)h[(size_t)bt * Dc + d] * w[d];
  out[bt] = acc;
}

// ---------------------------------------------------------------------------
// Tiled WMMA GEMM: C[M,N] = A[M,K] * W[N,K]^T + bias0 + bias1
// Block tile 64(M) x 128(N), K chunked by 32 through LDS via async copies.
// Requirements: K % 32 == 0, M % 64 == 0, W has >= ceil(N/128)*128 rows.
// FRAG=true stores in C-fragment-native layout (N must be multiple of 16):
//   out[(((m_tile*(N/16) + n_tile)*32 + lane)*8 + r] -- contiguous per lane.
// ---------------------------------------------------------------------------
template <bool OUT_F16, bool FRAG>
__global__ __launch_bounds__(256)
void k_gemm(const half_t* __restrict__ A, const half_t* __restrict__ W,
            const float* __restrict__ b0, const float* __restrict__ b1,
            void* __restrict__ Cv, int M, int N, int K, int ldc) {
  __shared__ half_t As[64][48];
  __shared__ half_t Ws[128][48];
  const int tid = threadIdx.x, lane = tid & 31, wv = tid >> 5;
  const int n0 = blockIdx.x * 128, m0 = blockIdx.y * 64;
  const v8f zf = {0.f, 0.f, 0.f, 0.f, 0.f, 0.f, 0.f, 0.f};
  v8f acc[4] = {zf, zf, zf, zf};
  const int smy = (wv >> 1) * 16;
  const int nb  = (wv & 1) * 64;

  for (int kc = 0; kc < K; kc += 32) {
    {
      int row = tid >> 2, sub = tid & 3;
      async_ldg_b128(&As[row][sub * 8], A + (size_t)(m0 + row) * K + kc + sub * 8);
    }
#pragma unroll
    for (int s = 0; s < 2; ++s) {
      int s2 = tid + s * 256;
      int row = s2 >> 2, sub = s2 & 3;
      async_ldg_b128(&Ws[row][sub * 8], W + (size_t)(n0 + row) * K + kc + sub * 8);
    }
    wait_async0();
    __syncthreads();
    v16h aA = load_frag_rm(&As[0][0], 48, smy, 0, lane);
#pragma unroll
    for (int j = 0; j < 4; ++j) {
      v16h bW = load_frag_rm(&Ws[0][0], 48, nb + j * 16, 0, lane);
      acc[j] = wmma_f16(aA, bW, acc[j]);
    }
    __syncthreads();
  }

  const int colL = lane & 15, hoff = (lane >> 4) << 3;
#pragma unroll
  for (int j = 0; j < 4; ++j) {
    int n = n0 + nb + j * 16 + colL;
    if (!FRAG && n >= N) continue;
    float bb = (b0 ? b0[n] : 0.f) + (b1 ? b1[n] : 0.f);
    if (FRAG) {
      int mt  = (m0 + smy) >> 4;
      int ntj = ((n0 + nb) >> 4) + j;
      size_t base = (((size_t)mt * (N >> 4) + ntj) * 32 + lane) * 8;
      if (OUT_F16) {
        half_t* d = (half_t*)Cv + base;
#pragma unroll
        for (int r = 0; r < 8; ++r) d[r] = (half_t)(acc[j][r] + bb);
      } else {
        float* d = (float*)Cv + base;
#pragma unroll
        for (int r = 0; r < 8; ++r) d[r] = acc[j][r] + bb;
      }
    } else {
#pragma unroll
      for (int r = 0; r < 8; ++r) {
        int m = m0 + smy + r + hoff;
        float v = acc[j][r] + bb;
        if (OUT_F16) ((half_t*)Cv)[(size_t)m * ldc + n] = (half_t)v;
        else         ((float*)Cv)[(size_t)m * ldc + n]  = v;
      }
    }
  }
}

// ---------------------------------------------------------------------------
// Encoder LSTM scan, both directions in one launch (blockIdx.x = direction).
// xg is in fragment layout with M-rows (s*32+b): tile mt = 2s + rowtile.
// Whh streamed from L2 every step (no restrict: LICM must not hoist+spill).
// ---------------------------------------------------------------------------
__global__ __launch_bounds__(512)
void k_lstm_enc2(const float* __restrict__ xgF, const float* __restrict__ xgB,
                 const half_t* WhhF, const half_t* WhhB, half_t* enc) {
  const int dir = blockIdx.x;
  const float*  xg  = dir ? xgB : xgF;
  const half_t* Whh = dir ? WhhB : WhhF;
  const int enc_off = dir ? Hc : 0;
  __shared__ half_t hbuf[Bc][Hc];
  const int tid = threadIdx.x, lane = tid & 31, wv = tid >> 5;
  for (int i = tid; i < Bc * Hc; i += 512) ((half_t*)hbuf)[i] = (half_t)0.f;
  __syncthreads();
  const int jb = wv * 16;
  const int colL = lane & 15, hoff = (lane >> 4) << 3;
  float cst[2][8];
#pragma unroll
  for (int rt = 0; rt < 2; ++rt)
#pragma unroll
    for (int r = 0; r < 8; ++r) cst[rt][r] = 0.f;

  for (int step = 0; step < Lc; ++step) {
    const int s = dir ? (Lc - 1 - step) : step;
    v8f acc[4][2];
#pragma unroll
    for (int gi = 0; gi < 4; ++gi)
#pragma unroll
      for (int rt = 0; rt < 2; ++rt) {
        const float* src =
            xg + ((((size_t)(2 * s + rt)) * 64 + (gi * 16 + wv)) * 32 + lane) * 8;
#pragma unroll
        for (int r = 0; r < 8; ++r) acc[gi][rt][r] = src[r];
      }
    for (int kt = 0; kt < Hc / 32; ++kt) {
      const int k0 = kt * 32;
      v16h a0 = load_frag_rm(&hbuf[0][0], Hc, 0,  k0, lane);
      v16h a1 = load_frag_rm(&hbuf[0][0], Hc, 16, k0, lane);
#pragma unroll
      for (int gi = 0; gi < 4; ++gi) {
        v16h bw = load_frag_rm(Whh, Hc, gi * Hc + jb, k0, lane);
        acc[gi][0] = wmma_f16(a0, bw, acc[gi][0]);
        acc[gi][1] = wmma_f16(a1, bw, acc[gi][1]);
      }
    }
    float hr[2][8];
#pragma unroll
    for (int rt = 0; rt < 2; ++rt)
#pragma unroll
      for (int r = 0; r < 8; ++r) {
        float iv = sigm(acc[0][rt][r]);
        float fv = sigm(acc[1][rt][r]);
        float gv = tanhf(acc[2][rt][r]);
        float ov = sigm(acc[3][rt][r]);
        float cc = fv * cst[rt][r] + iv * gv;
        cst[rt][r] = cc;
        hr[rt][r] = ov * tanhf(cc);
      }
    __syncthreads();
#pragma unroll
    for (int rt = 0; rt < 2; ++rt)
#pragma unroll
      for (int r = 0; r < 8; ++r) {
        int m = rt * 16 + r + hoff;
        half_t hv = (half_t)hr[rt][r];
        hbuf[m][jb + colL] = hv;
        enc[((size_t)m * Lc + s) * ENCc + enc_off + jb + colL] = hv;
      }
    __syncthreads();
  }
}

// ---------------------------------------------------------------------------
// Decoder LSTM scan split across DGW workgroups. WG g owns hidden columns
// [g*64, g*64+64) (4 waves x 16 cols). xg in fragment layout with M-rows
// (t*32+b): tile mt = 2t + rowtile. h ping-pongs in global memory; one
// device barrier per step. Full 32KB h async-copied to LDS each step.
// ---------------------------------------------------------------------------
__global__ __launch_bounds__(128)
void k_lstm_dec(const half_t* __restrict__ xg, const half_t* Whh, half_t* hall,
                half_t* hstate, unsigned* bar) {
  __shared__ half_t hbuf[Bc][Dc];   // 32 KB
  const int tid = threadIdx.x, lane = tid & 31, wv = tid >> 5;
  const int jb = blockIdx.x * 64 + wv * 16;
  const int jbt = jb >> 4;
  const int colL = lane & 15, hoff = (lane >> 4) << 3;
  float cst[2][8];
#pragma unroll
  for (int rt = 0; rt < 2; ++rt)
#pragma unroll
    for (int r = 0; r < 8; ++r) cst[rt][r] = 0.f;

  for (int t = 0; t < Tc; ++t) {
    const half_t* hsrc = hstate + (size_t)(t & 1) * (Bc * Dc);
    half_t*       hdst = hstate + (size_t)((t + 1) & 1) * (Bc * Dc);
    for (int s = tid; s < (Bc * Dc) / 8; s += 128) {
      async_ldg_b128(&((half_t*)hbuf)[s * 8], hsrc + s * 8);
    }
    wait_async0();
    __syncthreads();
    v8f acc[4][2];
#pragma unroll
    for (int gi = 0; gi < 4; ++gi)
#pragma unroll
      for (int rt = 0; rt < 2; ++rt) {
        const half_t* src =
            xg + ((((size_t)(2 * t + rt)) * 128 + (gi * 32 + jbt)) * 32 + lane) * 8;
#pragma unroll
        for (int r = 0; r < 8; ++r) acc[gi][rt][r] = (float)src[r];
      }
    for (int kt = 0; kt < Dc / 32; ++kt) {
      const int k0 = kt * 32;
      v16h a0 = load_frag_rm(&hbuf[0][0], Dc, 0,  k0, lane);
      v16h a1 = load_frag_rm(&hbuf[0][0], Dc, 16, k0, lane);
#pragma unroll
      for (int gi = 0; gi < 4; ++gi) {
        v16h bw = load_frag_rm(Whh, Dc, gi * Dc + jb, k0, lane);
        acc[gi][0] = wmma_f16(a0, bw, acc[gi][0]);
        acc[gi][1] = wmma_f16(a1, bw, acc[gi][1]);
      }
    }
#pragma unroll
    for (int rt = 0; rt < 2; ++rt)
#pragma unroll
      for (int r = 0; r < 8; ++r) {
        float iv = sigm(acc[0][rt][r]);
        float fv = sigm(acc[1][rt][r]);
        float gv = tanhf(acc[2][rt][r]);
        float ov = sigm(acc[3][rt][r]);
        float cc = fv * cst[rt][r] + iv * gv;
        cst[rt][r] = cc;
        float hf = ov * tanhf(cc);
        int m = rt * 16 + r + hoff;
        half_t hv = (half_t)hf;
        hdst[(size_t)m * Dc + jb + colL] = hv;
        hall[((size_t)m * Tc + t) * Dc + jb + colL] = hv;
      }
    gbar(bar, bar + 1, DGW);
  }
}

// ---------------------------------------------------------------------------
// Fused flash attention: one block per (b, head). 8 waves, 16-query tiles,
// K async-preloaded to LDS, V from transposed Vt buffer, online softmax.
// ---------------------------------------------------------------------------
__global__ __launch_bounds__(256)
void k_attn(const half_t* __restrict__ Qp, const half_t* __restrict__ Kp,
            const half_t* __restrict__ Vt, half_t* __restrict__ ctx) {
  __shared__ half_t Kl[Lc][64];        // 32 KB, async preload
  __shared__ float  scr[8][16][32];
  __shared__ half_t pl[8][16][32];
  __shared__ float  stm[8][16], stdn[8][16], rsc[8][16];
  const int tid = threadIdx.x, lane = tid & 31, wv = tid >> 5;
  const int bh = blockIdx.x, b = bh >> 3, hh = bh & 7;
  const v8f zf = {0.f, 0.f, 0.f, 0.f, 0.f, 0.f, 0.f, 0.f};

  for (int s = tid; s < Lc * 8; s += 256) {
    int row = s >> 3, sub = s & 7;
    async_ldg_b128(&Kl[row][sub * 8],
                   Kp + ((size_t)(b * Lc + row)) * ENCc + hh * 64 + sub * 8);
  }
  wait_async0();
  __syncthreads();

  const int colL = lane & 15, hoff = (lane >> 4) << 3;
  const half_t* Vtb = Vt + (size_t)bh * 64 * Lc;   // [d][key]

  for (int qt = wv; qt < Tc / 16; qt += 8) {
    const int t0 = qt * 16;
    const half_t* Qbase = Qp + ((size_t)b * Tc + t0) * ENCc + hh * 64;
    v8f o0 = zf, o1 = zf, o2 = zf, o3 = zf;
    if (lane < 16) { stm[wv][lane] = -1e30f; stdn[wv][lane] = 0.f; }
    v16h aq0 = load_frag_rm(Qbase, ENCc, 0, 0,  lane);
    v16h aq1 = load_frag_rm(Qbase, ENCc, 0, 32, lane);

    for (int kc = 0; kc < Lc / 32; ++kc) {
      const int key0 = kc * 32;
      v8f s0 = zf, s1 = zf;
      v16h bk;
      bk = load_frag_rm(&Kl[0][0], 64, key0,      0,  lane); s0 = wmma_f16(aq0, bk, s0);
      bk = load_frag_rm(&Kl[0][0], 64, key0,      32, lane); s0 = wmma_f16(aq1, bk, s0);
      bk = load_frag_rm(&Kl[0][0], 64, key0 + 16, 0,  lane); s1 = wmma_f16(aq0, bk, s1);
      bk = load_frag_rm(&Kl[0][0], 64, key0 + 16, 32, lane); s1 = wmma_f16(aq1, bk, s1);
#pragma unroll
      for (int r = 0; r < 8; ++r) {
        int m = r + hoff;
        scr[wv][m][colL]      = s0[r] * 0.125f;   // 1/sqrt(64)
        scr[wv][m][16 + colL] = s1[r] * 0.125f;
      }
      __builtin_amdgcn_wave_barrier();
      if (lane < 16) {                             // per-row online softmax
        int row = lane;
        float mold = stm[wv][row], dold = stdn[wv][row];
        float cm = -1e30f;
        for (int n = 0; n < 32; ++n) cm = fmaxf(cm, scr[wv][row][n]);
        float mnew = fmaxf(mold, cm);
        float sc = __expf(mold - mnew);
        float dn = dold * sc;
        for (int n = 0; n < 32; ++n) {
          float p = __expf(scr[wv][row][n] - mnew);
          dn += p;
          pl[wv][row][n] = (half_t)p;
        }
        stm[wv][row] = mnew; stdn[wv][row] = dn; rsc[wv][row] = sc;
      }
      __builtin_amdgcn_wave_barrier();
#pragma unroll
      for (int r = 0; r < 8; ++r) {
        float f = rsc[wv][r + hoff];
        o0[r] *= f; o1[r] *= f; o2[r] *= f; o3[r] *= f;
      }
      v16h ap = load_frag_rm(&pl[wv][0][0], 32, 0, 0, lane);
      o0 = wmma_f16(ap, load_frag_rm(Vtb, Lc, 0,  key0, lane), o0);
      o1 = wmma_f16(ap, load_frag_rm(Vtb, Lc, 16, key0, lane), o1);
      o2 = wmma_f16(ap, load_frag_rm(Vtb, Lc, 32, key0, lane), o2);
      o3 = wmma_f16(ap, load_frag_rm(Vtb, Lc, 48, key0, lane), o3);
    }
#pragma unroll
    for (int r = 0; r < 8; ++r) {
      int m = r + hoff;
      float inv = 1.f / fmaxf(stdn[wv][m], 1e-20f);
      size_t rowo = ((size_t)b * Tc + t0 + m) * ENCc + hh * 64;
      ctx[rowo + 0  + colL] = (half_t)(o0[r] * inv);
      ctx[rowo + 16 + colL] = (half_t)(o1[r] * inv);
      ctx[rowo + 32 + colL] = (half_t)(o2[r] * inv);
      ctx[rowo + 48 + colL] = (half_t)(o3[r] * inv);
    }
  }
}

// ---------------------------------------------------------------------------
// Host-side orchestration
// ---------------------------------------------------------------------------
extern "C" void kernel_launch(void* const* d_in, const int* in_sizes, int n_in,
                              void* d_out, int out_size, void* d_ws, size_t ws_size,
                              hipStream_t stream) {
  const int*   text  = (const int*)  d_in[0];
  const float* melT  = (const float*)d_in[1];
  const float* emb   = (const float*)d_in[2];
  const float* WihF  = (const float*)d_in[3];
  const float* WhhF  = (const float*)d_in[4];
  const float* bihF  = (const float*)d_in[5];
  const float* bhhF  = (const float*)d_in[6];
  const float* WihB  = (const float*)d_in[7];
  const float* WhhB  = (const float*)d_in[8];
  const float* bihB  = (const float*)d_in[9];
  const float* bhhB  = (const float*)d_in[10];
  const float* Wip   = (const float*)d_in[11];
  const float* bip   = (const float*)d_in[12];
  const float* inW   = (const float*)d_in[13];
  const float* inB   = (const float*)d_in[14];
  const float* outW  = (const float*)d_in[15];
  const float* outB  = (const float*)d_in[16];
  const float* WihD  = (const float*)d_in[17];
  const float* WhhD  = (const float*)d_in[18];
  const float* bihD  = (const float*)d_in[19];
  const float* bhhD  = (const float*)d_in[20];
  const float* Wmel  = (const float*)d_in[21];
  const float* bmel  = (const float*)d_in[22];
  const float* Wstop = (const float*)d_in[23];
  const float* bstop = (const float*)d_in[24];
  float* outp = (float*)d_out;

  char* ws = (char*)d_ws;
  size_t off = 0;
  auto alloc = [&](size_t bytes) -> char* {
    char* p = ws + off;
    off += (bytes + 255) & ~(size_t)255;
    return p;
  };

  half_t* hWihF = (half_t*)alloc((size_t)4 * Hc * Ec * 2);
  half_t* hWhhF = (half_t*)alloc((size_t)4 * Hc * Hc * 2);
  half_t* hWihB = (half_t*)alloc((size_t)4 * Hc * Ec * 2);
  half_t* hWhhB = (half_t*)alloc((size_t)4 * Hc * Hc * 2);
  half_t* hWip  = (half_t*)alloc((size_t)ENCc * KQP * 2);
  half_t* hQKV  = (half_t*)alloc((size_t)3 * ENCc * ENCc * 2);
  half_t* hOutW = (half_t*)alloc((size_t)ENCc * ENCc * 2);
  half_t* hWihD = (half_t*)alloc((size_t)4 * Dc * KDP * 2);
  half_t* hWhhD = (half_t*)alloc((size_t)4 * Dc * Dc * 2);
  half_t* hWmel = (half_t*)alloc((size_t)128 * Dc * 2);
  half_t* x16  = (half_t*)alloc((size_t)Bc * Lc * Ec * 2);
  float*  xgF  = (float*) alloc((size_t)Bc * Lc * 4 * Hc * 4);   // reused: attn-out
  float*  xgB  = (float*) alloc((size_t)Bc * Lc * 4 * Hc * 4);   // reused: lstm_in
  half_t* encB = (half_t*)alloc((size_t)Bc * Lc * ENCc * 2);
  half_t* decI = (half_t*)alloc((size_t)Bc * Tc * KQP * 2);
  half_t* qbuf = (half_t*)alloc((size_t)Bc * Tc * ENCc * 2);     // q, reused: ctx
  half_t* Qp   = (half_t*)alloc((size_t)Bc * Tc * ENCc * 2);
  half_t* Kp   = (half_t*)alloc((size_t)Bc * Lc * ENCc * 2);
  half_t* Vp   = (half_t*)alloc((size_t)Bc * Lc * ENCc * 2);
  half_t* Vt   = (half_t*)alloc((size_t)Bc * NHc * 64 * Lc * 2);
  half_t* xgD  = (half_t*)alloc((size_t)Bc * Tc * 4 * Dc * 2);
  half_t* hall = (half_t*)alloc((size_t)Bc * Tc * Dc * 2);
  half_t* hstate = (half_t*)alloc((size_t)2 * Bc * Dc * 2);
  unsigned* bar  = (unsigned*)alloc(256);
  half_t* attO = (half_t*)xgF;
  half_t* lin  = (half_t*)xgB;

  auto cast = [&](const float* s, half_t* d, int R, int K, int Rp, int Kp) {
    unsigned g = (unsigned)(((size_t)Rp * Kp + 255) / 256);
    k_cast_pad<<<dim3(g), dim3(256), 0, stream>>>(s, d, R, K, Rp, Kp);
  };
  auto gemm = [&](const half_t* A, const half_t* W, const float* b0, const float* b1,
                  void* C, int M, int N, int K, int ldc, bool f16out, bool frag) {
    dim3 grid((unsigned)((N + 127) / 128), (unsigned)(M / 64));
    if (f16out) {
      if (frag) k_gemm<true,  true ><<<grid, 256, 0, stream>>>(A, W, b0, b1, C, M, N, K, ldc);
      else      k_gemm<true,  false><<<grid, 256, 0, stream>>>(A, W, b0, b1, C, M, N, K, ldc);
    } else {
      if (frag) k_gemm<false, true ><<<grid, 256, 0, stream>>>(A, W, b0, b1, C, M, N, K, ldc);
      else      k_gemm<false, false><<<grid, 256, 0, stream>>>(A, W, b0, b1, C, M, N, K, ldc);
    }
  };

  // 1. weight casts (+ zero padding)
  cast(WihF, hWihF, 4 * Hc, Ec, 4 * Hc, Ec);
  cast(WhhF, hWhhF, 4 * Hc, Hc, 4 * Hc, Hc);
  cast(WihB, hWihB, 4 * Hc, Ec, 4 * Hc, Ec);
  cast(WhhB, hWhhB, 4 * Hc, Hc, 4 * Hc, Hc);
  cast(Wip,  hWip,  ENCc, Mc, ENCc, KQP);
  cast(inW,  hQKV,  3 * ENCc, ENCc, 3 * ENCc, ENCc);
  cast(outW, hOutW, ENCc, ENCc, ENCc, ENCc);
  cast(WihD, hWihD, 4 * Dc, ENCc + Mc, 4 * Dc, KDP);
  cast(WhhD, hWhhD, 4 * Dc, Dc, 4 * Dc, Dc);
  cast(Wmel, hWmel, Mc, Dc, 128, Dc);

  // 2. embedding (time-major rows s*32+b)
  k_embed<<<dim3(Bc * Lc), dim3(Ec), 0, stream>>>(text, emb, x16);

  // 3. encoder gate GEMMs (fragment-layout f32 out) + dual-direction scan
  gemm(x16, hWihF, bihF, bhhF, xgF, Bc * Lc, 4 * Hc, Ec, 0, false, true);
  gemm(x16, hWihB, bihB, bhhB, xgB, Bc * Lc, 4 * Hc, Ec, 0, false, true);
  k_lstm_enc2<<<dim3(2), dim3(512), 0, stream>>>(xgF, xgB, hWhhF, hWhhB, encB);

  // 4. decoder inputs + attention projections
  k_decin<<<dim3((Bc * Tc * KQP + 255) / 256), dim3(256), 0, stream>>>(
      melT, decI, Bc * Tc * KQP);
  gemm(decI, hWip, bip, nullptr, qbuf, Bc * Tc, ENCc, KQP, ENCc, true, false);
  gemm(qbuf, hQKV,                           inB,            nullptr, Qp, Bc * Tc, ENCc, ENCc, ENCc, true, false);
  gemm(encB, hQKV + (size_t)ENCc * ENCc,     inB + ENCc,     nullptr, Kp, Bc * Lc, ENCc, ENCc, ENCc, true, false);
  gemm(encB, hQKV + (size_t)2 * ENCc * ENCc, inB + 2 * ENCc, nullptr, Vp, Bc * Lc, ENCc, ENCc, ENCc, true, false);
  k_vt<<<dim3((Bc * NHc * 64 * Lc + 255) / 256), dim3(256), 0, stream>>>(
      Vp, Vt, Bc * NHc * 64 * Lc);

  // 5. fused attention -> ctx (reuses q buffer)
  half_t* ctx = qbuf;
  k_attn<<<dim3(Bc * NHc), dim3(256), 0, stream>>>(Qp, Kp, Vt, ctx);

  // 6. out-projection, concat (time-major), decoder gate GEMM (frag f16 out)
  gemm(ctx, hOutW, outB, nullptr, attO, Bc * Tc, ENCc, ENCc, ENCc, true, false);
  k_cat<<<dim3((Bc * Tc * KDP + 255) / 256), dim3(256), 0, stream>>>(
      attO, decI, lin, Bc * Tc * KDP);
  gemm(lin, hWihD, bihD, bhhD, xgD, Bc * Tc, 4 * Dc, KDP, 0, true, true);

  // 7. decoder LSTM scan: zero h0 + barrier state, then 8-WG scan
  {
    int zwords = (int)(((size_t)2 * Bc * Dc * 2 + 256) / 4);
    k_zero<<<dim3((zwords + 255) / 256), dim3(256), 0, stream>>>(
        (unsigned*)hstate, zwords);
  }
  k_lstm_dec<<<dim3(DGW), dim3(128), 0, stream>>>(xgD, hWhhD, hall, hstate, bar);

  // 8. outputs: mel (GEMM, f32 straight into d_out) + stop (dot)
  gemm(hall, hWmel, bmel, nullptr, outp, Bc * Tc, Mc, Dc, Mc, false, false);
  k_stop<<<dim3((Bc * Tc + 255) / 256), dim3(256), 0, stream>>>(
      hall, Wstop, bstop, outp + (size_t)Bc * Tc * Mc);
}
